// LocalAttention_9723805958496
// MI455X (gfx1250) — matrix operand order repaired and checked by reading, not compiled, not run on hardware
//
#include <hip/hip_runtime.h>
#include <hip/hip_bf16.h>
#include <math.h>

// ---------------------------------------------------------------------------
// LocalAttention (channel attention over overlapping 320x320 patches) on
// MI455X / gfx1250.  Memory-bound (~11 GFLOP vs a few hundred MB), so this
// version fuses qkv-conv + depthwise-3x3 + gram reduction into one kernel so
// the 201MB qkv intermediate never touches HBM.  All GEMM-shaped math uses
// v_wmma_f32_16x16x32_bf16.
// ---------------------------------------------------------------------------

typedef __attribute__((ext_vector_type(16))) __bf16 v16bf;
typedef __attribute__((ext_vector_type(8)))  float  v8f;

#define HEADS  4
#define Himg   512
#define Wimg   512
#define NPIX   (Himg * Wimg)      // 262144

// fused-kernel tiling
#define TW      32                // output tile width
#define THt     8                 // output tile height
#define HTW     34                // halo width
#define HTH     10                // halo height
#define HPX     (HTW * HTH)       // 340 halo pixels
#define NT      22                // ceil(340/16) WMMA column tiles
#define QH_STR  356               // LDS stride for halo buffer (>= 352, padded)
#define DW_STR  264               // LDS stride for dw outputs (256 + pad)

static __device__ __forceinline__ v8f wmma_bf16(v16bf a, v16bf b, v8f c) {
  return __builtin_amdgcn_wmma_f32_16x16x32_bf16(false, a, false, b, (short)0, c,
                                                 false, false);
}

// ---------------------------------------------------------------------------
// Stage 1: compute one 64-channel group of qkv = Wqkv @ x for the halo tile,
// WMMA bf16, results to LDS (bf16).  Out-of-image halo pixels -> 0 (SAME pad).
// ---------------------------------------------------------------------------
static __device__ __forceinline__ void qkv_group_to_lds(
    const float* __restrict__ x, const float* __restrict__ wq, int gb,
    int y0, int x0, int wv, int lane, __bf16 (*qh)[QH_STR]) {
  const int n  = lane & 15;
  const int hi = lane >> 4;

  // A fragments for this group's 4 oc tiles (loaded once, reused per tile)
  v16bf A0[4], A1[4];
#pragma unroll
  for (int ot = 0; ot < 4; ++ot) {
    const int oc = gb + ot * 16 + n;
#pragma unroll
    for (int e = 0; e < 16; ++e) {
      const int k = hi * 16 + e;   // contiguous k order
      A0[ot][e] = (__bf16)wq[oc * 64 + k];
      A1[ot][e] = (__bf16)wq[oc * 64 + 32 + k];
    }
  }

  for (int tt = wv; tt < NT; tt += 8) {
    const int slot = tt * 16 + n;
    const int hy = slot / HTW, hx = slot % HTW;
    const int y = y0 - 1 + hy, xg = x0 - 1 + hx;
    const bool ok = (slot < HPX) && (y >= 0) && (y < Himg) && (xg >= 0) && (xg < Wimg);
    const size_t pix = (size_t)y * Wimg + xg;

    v16bf b0, b1;
#pragma unroll
    for (int e = 0; e < 16; ++e) {
      const int k = hi * 16 + e;
      b0[e] = ok ? (__bf16)x[(size_t)k * NPIX + pix] : (__bf16)0.f;
      b1[e] = ok ? (__bf16)x[(size_t)(k + 32) * NPIX + pix] : (__bf16)0.f;
    }
#pragma unroll
    for (int ot = 0; ot < 4; ++ot) {
      v8f acc = {};
      acc = wmma_bf16(A0[ot], b0, acc);
      acc = wmma_bf16(A1[ot], b1, acc);
#pragma unroll
      for (int r = 0; r < 8; ++r)
        qh[ot * 16 + r + hi * 8][slot] = (__bf16)acc[r];
    }
  }
}

// Branch-free 9-tap depthwise conv from the zero-padded LDS halo buffer.
// Output (r,xx) in tile -> taps at halo coords (r+dy, xx+dx), dy,dx in 0..2.
static __device__ __forceinline__ float conv9_lds(const __bf16 (*qh)[QH_STR],
                                                  int ch, int r, int xx,
                                                  const float* w9) {
  float acc = 0.f;
#pragma unroll
  for (int dy = 0; dy < 3; ++dy) {
    const int base = (r + dy) * HTW + xx;
#pragma unroll
    for (int dx = 0; dx < 3; ++dx)
      acc += w9[dy * 3 + dx] * (float)qh[ch][base + dx];
  }
  return acc;
}

// ---------------------------------------------------------------------------
// Fused kernel: qkv GEMM (WMMA) -> dw 3x3 -> regional grams (WMMA) + sumsq,
// v channels dw'd and stored.  Tile 8x32, never straddles region boundaries.
// ---------------------------------------------------------------------------
__global__ __launch_bounds__(256) void k_fused(
    const float* __restrict__ x,          // [64][NPIX]
    const float* __restrict__ wq,         // [192][64]
    const float* __restrict__ dww,        // [192][9]
    float* __restrict__ v,                // [64][NPIX]
    float* __restrict__ Greg,             // [9][HEADS][16][16]
    float* __restrict__ SQ) {             // [2][9][64]
  __shared__ __bf16 qh[64][QH_STR];       // qkv halo, one 64-ch group (45.6KB)
  __shared__ __bf16 dq[64][DW_STR];       // dw(q)  (33.8KB)
  __shared__ __bf16 dk[64][DW_STR];       // dw(k)  (33.8KB)
  __shared__ float  gsum[HEADS][16][16];
  __shared__ float  sqbuf[256];

  const int t    = threadIdx.x;
  const int lane = t & 31;
  const int wv   = t >> 5;
  const int x0   = blockIdx.x * TW;
  const int y0   = blockIdx.y * THt;
  const int ry   = y0 < 192 ? 0 : (y0 < 320 ? 1 : 2);
  const int rx   = x0 < 192 ? 0 : (x0 < 320 ? 1 : 2);
  const int reg  = ry * 3 + rx;

  for (int i = t; i < HEADS * 256; i += 256) ((float*)gsum)[i] = 0.f;
  __syncthreads();

  const int ch = t & 63;     // channel within group
  const int r0 = t >> 6;     // 0..3 (covers rows r0 and r0+4)

  // ================= Q group (channels 0..63) =================
  qkv_group_to_lds(x, wq, 0, y0, x0, wv, lane, qh);
  __syncthreads();
  {
    float w9[9];
#pragma unroll
    for (int i = 0; i < 9; ++i) w9[i] = dww[ch * 9 + i];
    float sq = 0.f;
#pragma unroll
    for (int rp = 0; rp < 2; ++rp) {
      const int rr = r0 + rp * 4;
      for (int xx = 0; xx < TW; ++xx) {
        const float d = conv9_lds(qh, ch, rr, xx, w9);
        sq += d * d;
        dq[ch][rr * TW + xx] = (__bf16)d;
      }
    }
    sqbuf[t] = sq;
  }
  __syncthreads();
  if (t < 64) {
    const float s = sqbuf[t] + sqbuf[t + 64] + sqbuf[t + 128] + sqbuf[t + 192];
    atomicAdd(&SQ[0 * 576 + reg * 64 + t], s);
  }
  __syncthreads();   // qh reuse below

  // ================= K group (channels 64..127) =================
  qkv_group_to_lds(x, wq, 64, y0, x0, wv, lane, qh);
  __syncthreads();
  {
    float w9[9];
#pragma unroll
    for (int i = 0; i < 9; ++i) w9[i] = dww[(64 + ch) * 9 + i];
    float sq = 0.f;
#pragma unroll
    for (int rp = 0; rp < 2; ++rp) {
      const int rr = r0 + rp * 4;
      for (int xx = 0; xx < TW; ++xx) {
        const float d = conv9_lds(qh, ch, rr, xx, w9);
        sq += d * d;
        dk[ch][rr * TW + xx] = (__bf16)d;
      }
    }
    sqbuf[t] = sq;
  }
  __syncthreads();
  if (t < 64) {
    const float s = sqbuf[t] + sqbuf[t + 64] + sqbuf[t + 128] + sqbuf[t + 192];
    atomicAdd(&SQ[576 + reg * 64 + t], s);
  }
  __syncthreads();

  // ================= regional gram via WMMA =================
  {
    const int n    = lane & 15;
    const int hi   = lane >> 4;
    const int head = wv >> 1;
    const int cb   = (wv & 1) * 128;      // this wave's 128-pixel K chunk
    v8f acc = {};
#pragma unroll
    for (int c4 = 0; c4 < 4; ++c4) {
      const int kb = cb + c4 * 32;
      v16bf a, b;
#pragma unroll
      for (int e = 0; e < 16; ++e) {
        a[e] = dq[head * 16 + n][kb + hi * 16 + e];
        b[e] = dk[head * 16 + n][kb + hi * 16 + e];
      }
      acc = wmma_bf16(a, b, acc);
    }
#pragma unroll
    for (int rr = 0; rr < 8; ++rr)
      atomicAdd(&gsum[head][rr + hi * 8][n], acc[rr]);
  }
  __syncthreads();

  // ================= V group (channels 128..191) =================
  qkv_group_to_lds(x, wq, 128, y0, x0, wv, lane, qh);
  __syncthreads();
  for (int p = wv; p < 64 * THt; p += 8) {     // (vch, row) pairs; lane = x
    const int vch = p >> 3;
    const int rr  = p & 7;
    float w9[9];
#pragma unroll
    for (int i = 0; i < 9; ++i) w9[i] = dww[(128 + vch) * 9 + i];
    const float d = conv9_lds(qh, vch, rr, lane, w9);
    v[(size_t)vch * NPIX + (y0 + rr) * Wimg + x0 + lane] = d;
  }

  // ================= flush regional grams =================
  for (int i = t; i < HEADS * 256; i += 256)
    atomicAdd(&Greg[reg * (HEADS * 256) + i], ((float*)gsum)[i]);
}

// ---------------------------------------------------------------------------
// Kernel 2 (tiny, 1 block): per-patch grams from regional sums, l2-normalize,
// temperature, softmax; average attn per region; fold proj -> Mreg[9][64][64].
// ---------------------------------------------------------------------------
__global__ __launch_bounds__(256) void k_mats(const float* __restrict__ Greg,
                                              const float* __restrict__ SQ,
                                              const float* __restrict__ temp,   // [4]
                                              const float* __restrict__ proj,   // [64][64]
                                              float* __restrict__ Mreg) {       // [9][64][64]
  __shared__ float attn[4][HEADS][16][16];
  const int t = threadIdx.x;
  {
    const int p  = t >> 6;          // patch 0..3
    const int hh = (t >> 4) & 3;    // head
    const int cc = t & 15;          // q channel within head
    const int pi = p >> 1, pj = p & 1;
    const int ryA = pi ? 1 : 0, ryB = pi ? 2 : 1;
    const int rxA = pj ? 1 : 0, rxB = pj ? 2 : 1;

    float sq = 0.f;
    for (int a = ryA; a <= ryB; ++a)
      for (int b = rxA; b <= rxB; ++b)
        sq += SQ[0 * 576 + (a * 3 + b) * 64 + hh * 16 + cc];
    const float nq = fmaxf(sqrtf(sq), 1e-12f);
    const float T  = temp[hh];

    float row[16];
    float mx = -3.4e38f;
#pragma unroll
    for (int dd = 0; dd < 16; ++dd) {
      float g = 0.f, sk = 0.f;
      for (int a = ryA; a <= ryB; ++a)
        for (int b = rxA; b <= rxB; ++b) {
          const int rg = a * 3 + b;
          g  += Greg[rg * 1024 + hh * 256 + cc * 16 + dd];
          sk += SQ[576 + rg * 64 + hh * 16 + dd];
        }
      const float nk = fmaxf(sqrtf(sk), 1e-12f);
      const float l  = g / (nq * nk) * T;
      row[dd] = l;
      mx = fmaxf(mx, l);
    }
    float s = 0.f;
#pragma unroll
    for (int dd = 0; dd < 16; ++dd) { row[dd] = expf(row[dd] - mx); s += row[dd]; }
    const float inv = 1.f / s;
#pragma unroll
    for (int dd = 0; dd < 16; ++dd) attn[p][hh][cc][dd] = row[dd] * inv;
  }
  __syncthreads();

  // Mreg[r][o][h*16+dd] = sum_cc proj[o][h*16+cc] * avg_patches attn[cc][dd]
  for (int idx = t; idx < 9 * 64 * 64; idx += 256) {
    const int r  = idx >> 12;
    const int o  = (idx >> 6) & 63;
    const int cv = idx & 63;
    const int hh = cv >> 4, dd = cv & 15;
    const int ry = r / 3, rx = r % 3;
    const int piA = (ry == 2) ? 1 : 0, piB = (ry == 0) ? 0 : 1;
    const int pjA = (rx == 2) ? 1 : 0, pjB = (rx == 0) ? 0 : 1;
    const float invc = 1.f / (float)((piB - piA + 1) * (pjB - pjA + 1));
    float acc = 0.f;
#pragma unroll
    for (int cc = 0; cc < 16; ++cc) {
      float am = 0.f;
      for (int pi = piA; pi <= piB; ++pi)
        for (int pj = pjA; pj <= pjB; ++pj)
          am += attn[pi * 2 + pj][hh][cc][dd];
      acc += proj[o * 64 + hh * 16 + cc] * (am * invc);
    }
    Mreg[idx] = acc;
  }
}

// ---------------------------------------------------------------------------
// Kernel 3: out = Mreg[region(y,x)] (64x64) @ v per pixel, WMMA bf16.
// ---------------------------------------------------------------------------
__global__ __launch_bounds__(256) void k_out(const float* __restrict__ v,
                                             const float* __restrict__ Mreg,
                                             float* __restrict__ out) {
  const int wave = blockIdx.x * 8 + (threadIdx.x >> 5);
  const int lane = threadIdx.x & 31;
  const int n    = lane & 15;
  const int hi   = lane >> 4;
  const int pix0 = wave * 16;
  const int y    = pix0 >> 9;
  const int xx   = pix0 & 511;
  const int ry   = y  < 192 ? 0 : (y  < 320 ? 1 : 2);
  const int rx   = xx < 192 ? 0 : (xx < 320 ? 1 : 2);
  const float* M = Mreg + (ry * 3 + rx) * 4096;
  const int pix  = pix0 + n;

  v16bf b0, b1;
#pragma unroll
  for (int e = 0; e < 16; ++e) {
    const int k = hi * 16 + e;
    b0[e] = (__bf16)v[(size_t)k * NPIX + pix];
    b1[e] = (__bf16)v[(size_t)(k + 32) * NPIX + pix];
  }
#pragma unroll
  for (int ot = 0; ot < 4; ++ot) {
    const int o = ot * 16 + n;
    v16bf a0, a1;
#pragma unroll
    for (int e = 0; e < 16; ++e) {
      const int k = hi * 16 + e;
      a0[e] = (__bf16)M[o * 64 + k];
      a1[e] = (__bf16)M[o * 64 + 32 + k];
    }
    v8f acc = {};
    acc = wmma_bf16(a0, b0, acc);
    acc = wmma_bf16(a1, b1, acc);
#pragma unroll
    for (int r = 0; r < 8; ++r)
      out[(size_t)(ot * 16 + r + hi * 8) * NPIX + pix] = acc[r];
  }
}

// ---------------------------------------------------------------------------
// Launch
// ---------------------------------------------------------------------------
extern "C" void kernel_launch(void* const* d_in, const int* in_sizes, int n_in,
                              void* d_out, int out_size, void* d_ws, size_t ws_size,
                              hipStream_t stream) {
  const float* x     = (const float*)d_in[0];   // [64][512][512]
  const float* qkv_w = (const float*)d_in[1];   // [192][64]
  const float* dw_w  = (const float*)d_in[2];   // [192][9]
  const float* proj  = (const float*)d_in[3];   // [64][64]
  const float* temp  = (const float*)d_in[4];   // [4]
  float* out = (float*)d_out;                   // [64][512][512]

  float* ws   = (float*)d_ws;
  float* vbuf = ws;                              // 64*NPIX = 16,777,216 f
  float* Greg = vbuf + (size_t)64 * NPIX;        // 9*4*16*16 = 9,216 f
  float* SQ   = Greg + 9 * 1024;                 // 2*9*64    = 1,152 f
  float* Mreg = SQ   + 2 * 9 * 64;               // 9*64*64   = 36,864 f

  // zero the accumulation buffers (Greg + SQ are contiguous)
  hipMemsetAsync(Greg, 0, (9 * 1024 + 2 * 9 * 64) * sizeof(float), stream);

  k_fused<<<dim3(Wimg / TW, Himg / THt), 256, 0, stream>>>(x, qkv_w, dw_w,
                                                           vbuf, Greg, SQ);
  k_mats<<<1, 256, 0, stream>>>(Greg, SQ, temp, proj, Mreg);
  k_out<<<NPIX / 128, 256, 0, stream>>>(vbuf, Mreg, out);
}
